// ConstrainedDiffPool_4363686773147
// MI455X (gfx1250) — compile-verified, matrix operand order checked by reading, not run on hardware
//
#include <hip/hip_runtime.h>
#include <hip/hip_bf16.h>
#include <math.h>

typedef __attribute__((ext_vector_type(16))) __bf16 v16bf;
typedef __attribute__((ext_vector_type(8)))  __bf16 v8bf;
typedef __attribute__((ext_vector_type(8)))  float  v8f;

#define NN 6144
#define DD 256
#define KK 20
#define KP 32
#define EE 98304
#define TGT (6144.0f/20.0f)

// ---- 16-byte vector fragment load: elements [0..7] and [16..23] from p ----
__device__ __forceinline__ v16bf load_frag(const __bf16* __restrict__ p) {
  v8bf lo = *(const v8bf*)p;
  v8bf hi = *(const v8bf*)(p + 16);
  return __builtin_shufflevector(lo, hi, 0,1,2,3,4,5,6,7,8,9,10,11,12,13,14,15);
}

// ---------------- conversion (fp32 -> bf16, column zero-padding) ----------
__global__ void k_cvt(const float* __restrict__ src, __bf16* __restrict__ dst,
                      int rows, int cols, int colsPad) {
  int idx = blockIdx.x * blockDim.x + threadIdx.x;
  int tot = rows * colsPad;
  if (idx >= tot) return;
  int r = idx / colsPad, c = idx % colsPad;
  float v = (c < cols) ? src[(size_t)r * cols + c] : 0.0f;
  dst[idx] = (__bf16)v;
}

// ---- transpose-convert weights: src fp32 [K x Nsrc] -> dst bf16 [Npad x K]
__global__ void k_cvtT(const float* __restrict__ src, __bf16* __restrict__ dst,
                       int K, int Nsrc, int Npad) {
  int idx = blockIdx.x * blockDim.x + threadIdx.x;
  int tot = Npad * K;
  if (idx >= tot) return;
  int n = idx / K, k = idx % K;
  float v = (n < Nsrc) ? src[(size_t)k * Nsrc + n] : 0.0f;
  dst[idx] = (__bf16)v;
}

// ---- S (fp32, ld KP) -> S16 [NN x KP] and S16T [KP x NN] bf16 ------------
__global__ void k_cvtS(const float* __restrict__ S, __bf16* __restrict__ S16,
                       __bf16* __restrict__ S16T) {
  int idx = blockIdx.x * blockDim.x + threadIdx.x;
  if (idx >= NN * KP) return;
  int r = idx / KP, c = idx % KP;
  __bf16 v = (__bf16)S[idx];
  S16[idx] = v;
  S16T[(size_t)c * NN + r] = v;
}

// ---------------- bf16 WMMA GEMM, 16x32 output tile per wave --------------
// A row-major [M x KD] (lda=LDA), BT row-major [Nfull x KD] (ld=LDB: B transposed)
// C = act(A@B + bias); Cf fp32 (ld=LDC), Cb bf16 (normal ld=LDC, or transposed
// with ld=ldcb when cbT!=0).
template<int KD, int LDA, int LDB, int LDC, int RELU>
__global__ void k_wmma_gemm(const __bf16* __restrict__ A, const __bf16* __restrict__ BT,
                            const float* __restrict__ bias,
                            float* __restrict__ Cf, __bf16* __restrict__ Cb,
                            int cbT, int ldcb, int M, int Nfull, int nvalid) {
  const int lane = threadIdx.x & 31;
  const int r    = lane & 15;
  const int half = lane >> 4;
  const int m0 = blockIdx.y * 16;
  const int n0 = blockIdx.x * 32;
  v8f acc0 = {0.f,0.f,0.f,0.f,0.f,0.f,0.f,0.f};
  v8f acc1 = {0.f,0.f,0.f,0.f,0.f,0.f,0.f,0.f};
  const __bf16* arow  = A  + (size_t)(m0 + r) * LDA + half * 8;
  const __bf16* b0row = BT + (size_t)(n0 + r) * LDB + half * 8;
  const __bf16* b1row = BT + (size_t)(n0 + 16 + r) * LDB + half * 8;
  for (int k0 = 0; k0 < KD; k0 += 32) {
    v16bf a  = load_frag(arow + k0);
    v16bf b0 = load_frag(b0row + k0);
    v16bf b1 = load_frag(b1row + k0);
    if (k0 + 32 < KD) {
      __builtin_prefetch(arow + k0 + 32, 0, 0);
      __builtin_prefetch(b0row + k0 + 32, 0, 0);
      __builtin_prefetch(b1row + k0 + 32, 0, 0);
    }
    acc0 = __builtin_amdgcn_wmma_f32_16x16x32_bf16(false, a, false, b0, (short)0, acc0, false, false);
    acc1 = __builtin_amdgcn_wmma_f32_16x16x32_bf16(false, a, false, b1, (short)0, acc1, false, false);
  }
  const int rowbase = m0 + half * 8;
  #pragma unroll
  for (int t = 0; t < 2; ++t) {
    const int col = n0 + t * 16 + r;
    v8f acc = t ? acc1 : acc0;
    if (col >= Nfull) continue;
    #pragma unroll
    for (int rr = 0; rr < 8; ++rr) {
      int row = rowbase + rr;
      if (row >= M) continue;
      float v = acc[rr];
      if (col < nvalid) {
        if (bias) v += bias[col];
        if (RELU) v = fmaxf(v, 0.0f);
      } else {
        v = 0.0f;
      }
      if (Cf) Cf[(size_t)row * LDC + col] = v;
      if (Cb) {
        if (cbT) Cb[(size_t)col * ldcb + row] = (__bf16)v;
        else     Cb[(size_t)row * LDC + col] = (__bf16)v;
      }
    }
  }
}

// ------- N^2 link-loss: p = clip(S@S^T), accumulate log1p(-p) --------------
// Exploits symmetry of p: only tiles with col-range >= row-range do work;
// strictly-upper elements weighted 2x, diagonal 1x, lower 0x.
__global__ void k_ptile(const __bf16* __restrict__ S16, float* __restrict__ partials) {
  const int lane = threadIdx.x & 31;
  const int m0 = blockIdx.y * 16;
  const int n0 = blockIdx.x * 32;
  if (n0 + 32 <= m0) {              // tile entirely below the diagonal
    if (lane == 0) partials[(size_t)blockIdx.y * gridDim.x + blockIdx.x] = 0.f;
    return;
  }
  const int r    = lane & 15;
  const int half = lane >> 4;
  const int kb = half * 8;
  v16bf a  = load_frag(S16 + (size_t)(m0 + r) * KP + kb);
  v16bf b0 = load_frag(S16 + (size_t)(n0 + r) * KP + kb);
  v16bf b1 = load_frag(S16 + (size_t)(n0 + 16 + r) * KP + kb);
  v8f z = {0.f,0.f,0.f,0.f,0.f,0.f,0.f,0.f};
  v8f acc0 = __builtin_amdgcn_wmma_f32_16x16x32_bf16(false, a, false, b0, (short)0, z, false, false);
  v8f acc1 = __builtin_amdgcn_wmma_f32_16x16x32_bf16(false, a, false, b1, (short)0, z, false, false);
  const int rowbase = m0 + half * 8;
  float s = 0.f;
  #pragma unroll
  for (int t = 0; t < 2; ++t) {
    const int col = n0 + t * 16 + r;
    v8f acc = t ? acc1 : acc0;
    #pragma unroll
    for (int rr = 0; rr < 8; ++rr) {
      int row = rowbase + rr;
      float w = (col > row) ? 2.f : ((col == row) ? 1.f : 0.f);
      float p = fminf(fmaxf(acc[rr], 0.f), 1.f);
      // log1p(-p) == log(1-p); clamp to -100 BEFORE weighting (p==1 -> -inf -> -100)
      s += w * fmaxf(__logf(1.0f - p), -100.0f);
    }
  }
  #pragma unroll
  for (int off = 16; off > 0; off >>= 1) s += __shfl_xor(s, off, 32);
  if (lane == 0) partials[(size_t)blockIdx.y * gridDim.x + blockIdx.x] = s;
}

// ---------------- row softmax (optionally with colsum logit adjustment) ---
__global__ void k_softmax_rows(const float* __restrict__ L, const float* __restrict__ cs,
                               float* __restrict__ out) {
  int row = blockIdx.x * blockDim.x + threadIdx.x;
  if (row >= NN) return;
  float l[KK];
  #pragma unroll
  for (int j = 0; j < KK; ++j) {
    float v = L[(size_t)row * KP + j];
    if (cs) v -= 0.1f * (cs[j] - TGT) / TGT;
    l[j] = v;
  }
  float m = -3.402823466e38f;
  #pragma unroll
  for (int j = 0; j < KK; ++j) m = fmaxf(m, l[j]);
  float s = 0.f;
  #pragma unroll
  for (int j = 0; j < KK; ++j) { l[j] = expf(l[j] - m); s += l[j]; }
  float inv = 1.f / s;
  #pragma unroll
  for (int j = 0; j < KK; ++j) out[(size_t)row * KP + j] = l[j] * inv;
  #pragma unroll
  for (int j = KK; j < KP; ++j) out[(size_t)row * KP + j] = 0.f;
}

// ---------------- S_adj = softmax(log(S+eps) + pen) -----------------------
__global__ void k_sadj(const float* __restrict__ Sf, const float* __restrict__ pen,
                       float* __restrict__ Sadj) {
  int row = blockIdx.x * blockDim.x + threadIdx.x;
  if (row >= NN) return;
  float l[KK];
  #pragma unroll
  for (int j = 0; j < KK; ++j) l[j] = logf(Sf[(size_t)row * KP + j] + 1e-8f) + pen[j];
  float m = -3.402823466e38f;
  #pragma unroll
  for (int j = 0; j < KK; ++j) m = fmaxf(m, l[j]);
  float s = 0.f;
  #pragma unroll
  for (int j = 0; j < KK; ++j) { l[j] = expf(l[j] - m); s += l[j]; }
  float inv = 1.f / s;
  #pragma unroll
  for (int j = 0; j < KK; ++j) Sadj[(size_t)row * KP + j] = l[j] * inv;
  #pragma unroll
  for (int j = KK; j < KP; ++j) Sadj[(size_t)row * KP + j] = 0.f;
}

// ---------------- deterministic column sums (20 blocks) -------------------
__global__ void k_colsum(const float* __restrict__ S, float* __restrict__ cs) {
  __shared__ float red[256];
  int k = blockIdx.x, tid = threadIdx.x;
  float s = 0.f;
  for (int row = tid; row < NN; row += 256) s += S[(size_t)row * KP + k];
  red[tid] = s; __syncthreads();
  for (int off = 128; off > 0; off >>= 1) { if (tid < off) red[tid] += red[tid + off]; __syncthreads(); }
  if (tid == 0) cs[k] = red[0];
}

// ---------------- size-constraint penalty + flags -------------------------
__global__ void k_pen(const float* __restrict__ cs, float* __restrict__ pen,
                      int* __restrict__ vo, int* __restrict__ flag) {
  __shared__ int any;
  int t = threadIdx.x;
  if (t == 0) any = 0;
  __syncthreads();
  if (t < KK) {
    float c = cs[t];
    float p = 0.f;
    if (c > 20.0f) p = -logf(c / 20.0f + 1e-8f);
    else if (c > 0.0f && c < 3.0f) p = -0.5f * logf(c / 3.0f + 1e-8f);
    pen[t] = p;
    vo[t] = (c > 30.0f) ? 1 : 0;
    if (p != 0.f) atomicOr(&any, 1);
  }
  __syncthreads();
  if (t == 0) flag[0] = any;
}

// ---------------- block reduce helper (1024 threads) ----------------------
__device__ __forceinline__ unsigned blockReduceCount(unsigned v, unsigned* red) {
  int tid = threadIdx.x;
  red[tid] = v; __syncthreads();
  for (int off = 512; off > 0; off >>= 1) { if (tid < off) red[tid] += red[tid + off]; __syncthreads(); }
  unsigned r = red[0]; __syncthreads();
  return r;
}

// ---------------- quantile(0.7) threshold + zero weak + row renorm --------
__global__ void k_quant_zero(float* __restrict__ Sadj, const int* __restrict__ vo, int k) {
  __shared__ unsigned red[1024];
  if (!vo[k]) return;
  const int tid = threadIdx.x;
  const int row0 = tid * 6;
  unsigned keys[6];
  #pragma unroll
  for (int i = 0; i < 6; ++i) keys[i] = __float_as_uint(Sadj[(size_t)(row0 + i) * KP + k]);
  unsigned res[2];
  for (int s = 0; s < 2; ++s) {
    unsigned need = 4301u + (unsigned)s;  // order stats 4300, 4301 (0-indexed)
    unsigned lo = 0u, hi = 0xFFFFFFFFu;
    while (lo < hi) {
      unsigned mid = lo + ((hi - lo) >> 1);
      unsigned c = 0;
      #pragma unroll
      for (int i = 0; i < 6; ++i) c += (keys[i] <= mid) ? 1u : 0u;
      unsigned tot = blockReduceCount(c, red);
      if (tot >= need) hi = mid; else lo = mid + 1u;
    }
    res[s] = lo;
  }
  float v0 = __uint_as_float(res[0]), v1 = __uint_as_float(res[1]);
  float pos = 0.7f * (float)(NN - 1);
  float fl = floorf(pos), frac = pos - fl;
  float thr = v0 * (1.0f - frac) + v1 * frac;
  for (int i = 0; i < 6; ++i) {
    int row = row0 + i;
    float val = __uint_as_float(keys[i]);
    if (val < thr) {
      float* rw = Sadj + (size_t)row * KP;
      rw[k] = 0.0f;
      float rs = 0.f;
      #pragma unroll
      for (int j = 0; j < KK; ++j) rs += fabsf(rw[j]);
      rs = fmaxf(rs, 1e-12f);
      #pragma unroll
      for (int j = 0; j < KK; ++j) rw[j] /= rs;
    }
  }
}

// ---------------- hard rebalance, one column (stable tie handling) --------
__global__ void k_rebalance(float* __restrict__ S, const float* __restrict__ cs, int k) {
  __shared__ unsigned red[1024];
  __shared__ unsigned scanArr[1024];
  const int tid = threadIdx.x;
  float csk = cs[k];
  float excess = csk - 20.0f;
  if (!(csk > 20.0f && excess > 0.0f)) return;
  int num = (int)ceilf(excess) + 5;
  if (num > NN) num = NN;
  const int row0 = tid * 6;
  unsigned keys[6];
  #pragma unroll
  for (int i = 0; i < 6; ++i) keys[i] = __float_as_uint(S[(size_t)(row0 + i) * KP + k]);
  unsigned lo = 0u, hi = 0xFFFFFFFFu;
  while (lo < hi) {
    unsigned mid = lo + ((hi - lo) >> 1);
    unsigned c = 0;
    #pragma unroll
    for (int i = 0; i < 6; ++i) c += (keys[i] <= mid) ? 1u : 0u;
    unsigned tot = blockReduceCount(c, red);
    if (tot >= (unsigned)num) hi = mid; else lo = mid + 1u;
  }
  unsigned tstar = lo;
  unsigned cl = 0;
  #pragma unroll
  for (int i = 0; i < 6; ++i) cl += (keys[i] < tstar) ? 1u : 0u;
  unsigned c_less = blockReduceCount(cl, red);
  int needed = num - (int)c_less;
  unsigned ct = 0;
  #pragma unroll
  for (int i = 0; i < 6; ++i) ct += (keys[i] == tstar) ? 1u : 0u;
  scanArr[tid] = ct; __syncthreads();
  if (tid == 0) {
    unsigned run = 0;
    for (int i = 0; i < 1024; ++i) { unsigned t = scanArr[i]; scanArr[i] = run; run += t; }
  }
  __syncthreads();
  unsigned trk = scanArr[tid];
  float w[KK]; float wsum = 0.f;
  #pragma unroll
  for (int j = 0; j < KK; ++j) { float av = 20.0f - cs[j]; av = av > 0.f ? av : 0.f; w[j] = av; wsum += av; }
  wsum += 1e-8f;
  #pragma unroll
  for (int j = 0; j < KK; ++j) w[j] /= wsum;
  for (int i = 0; i < 6; ++i) {
    int row = row0 + i;
    bool weak;
    if (keys[i] < tstar) weak = true;
    else if (keys[i] == tstar) { weak = ((int)trk < needed); trk++; }
    else weak = false;
    if (weak) {
      float* rw = S + (size_t)row * KP;
      rw[k] = 1e-8f;
      const float colk = 1e-8f;
      #pragma unroll
      for (int j = 0; j < KK; ++j) rw[j] += w[j] * colk;
      float rs = 0.f;
      #pragma unroll
      for (int j = 0; j < KK; ++j) rs += fabsf(rw[j]);
      rs = fmaxf(rs, 1e-12f);
      #pragma unroll
      for (int j = 0; j < KK; ++j) rw[j] /= rs;
    }
  }
}

__global__ void k_select(const float* __restrict__ A, const float* __restrict__ B,
                         const int* __restrict__ flag, float* __restrict__ out, int n) {
  int i = blockIdx.x * blockDim.x + threadIdx.x;
  if (i < n) out[i] = flag[0] ? A[i] : B[i];
}

__global__ void k_zero32(unsigned* __restrict__ p, int n) {
  int i = blockIdx.x * blockDim.x + threadIdx.x;
  if (i < n) p[i] = 0u;
}

// ---------------- edges: dedup + adj_pooled partials + edge loss terms ----
__global__ void k_edges(const int* __restrict__ ei, const float* __restrict__ S,
                        unsigned* __restrict__ mask, float* __restrict__ edgeT,
                        float* __restrict__ blkAdj) {
  __shared__ float tile[KK * KK];
  const int tid = threadIdx.x;
  for (int i = tid; i < KK * KK; i += 256) tile[i] = 0.f;
  __syncthreads();
  const int e = blockIdx.x * 256 + tid;
  float term = 0.f;
  if (e < EE) {
    int u = ei[e], v = ei[EE + e];
    unsigned pos = (unsigned)u * (unsigned)NN + (unsigned)v;
    unsigned bit = 1u << (pos & 31u);
    unsigned old = atomicOr(&mask[pos >> 5], bit);
    if (!(old & bit)) {  // unique-edge winner (scatter-set dedup semantics)
      float su[KK], sv[KK], p = 0.f;
      #pragma unroll
      for (int a = 0; a < KK; ++a) {
        su[a] = S[(size_t)u * KP + a];
        sv[a] = S[(size_t)v * KP + a];
        p += su[a] * sv[a];
      }
      p = fminf(fmaxf(p, 0.f), 1.f);
      term = fmaxf(logf(p), -100.f) - fmaxf(log1pf(-p), -100.f);
      #pragma unroll
      for (int a = 0; a < KK; ++a)
        #pragma unroll
        for (int b = 0; b < KK; ++b)
          atomicAdd(&tile[a * KK + b], su[a] * sv[b]);
    }
    edgeT[e] = term;
  }
  __syncthreads();
  for (int i = tid; i < KK * KK; i += 256) blkAdj[(size_t)blockIdx.x * (KK * KK) + i] = tile[i];
}

__global__ void k_adj_reduce(const float* __restrict__ blkAdj, float* __restrict__ out) {
  int i = threadIdx.x;
  if (i < KK * KK) {
    float s = 0.f;
    for (int b = 0; b < EE / 256; ++b) s += blkAdj[(size_t)b * (KK * KK) + i];
    out[i] = s;
  }
}

__global__ void k_reduce_sum(const float* __restrict__ in, int n, float* __restrict__ out) {
  __shared__ float red[1024];
  int tid = threadIdx.x;
  float s = 0.f;
  for (int i = tid; i < n; i += 1024) s += in[i];
  red[tid] = s; __syncthreads();
  for (int off = 512; off > 0; off >>= 1) { if (tid < off) red[tid] += red[tid + off]; __syncthreads(); }
  if (tid == 0) out[0] = red[0];
}

__global__ void k_entropy(const float* __restrict__ S, float* __restrict__ out) {
  __shared__ float red[1024];
  int tid = threadIdx.x;
  float s = 0.f;
  for (int row = tid; row < NN; row += 1024) {
    float r = 0.f;
    #pragma unroll
    for (int j = 0; j < KK; ++j) { float v = S[(size_t)row * KP + j]; r += v * logf(v + 1e-8f); }
    s += r;
  }
  red[tid] = s; __syncthreads();
  for (int off = 512; off > 0; off >>= 1) { if (tid < off) red[tid] += red[tid + off]; __syncthreads(); }
  if (tid == 0) out[0] = red[0];
}

__global__ void k_scopy(const float* __restrict__ S, float* __restrict__ out) {
  int i = blockIdx.x * blockDim.x + threadIdx.x;
  if (i < NN * KK) {
    int r = i / KK, c = i % KK;
    out[i] = S[(size_t)r * KP + c];
  }
}

__global__ void k_final(const float* __restrict__ sc, const float* __restrict__ csF,
                        float* __restrict__ aux) {
  if (threadIdx.x == 0 && blockIdx.x == 0) {
    float link = -(sc[0] + sc[1]) / ((float)NN * (float)NN);
    float ent  = -sc[2] / (float)NN;
    float bal = 0.f;
    for (int j = 0; j < KK; ++j) bal += fabsf(csF[j] - TGT);
    bal = (bal / (float)KK) / TGT;
    aux[0] = link + 0.1f * ent + 0.5f * bal;
  }
}

// ===========================================================================
extern "C" void kernel_launch(void* const* d_in, const int* in_sizes, int n_in,
                              void* d_out, int out_size, void* d_ws, size_t ws_size,
                              hipStream_t stream) {
  (void)in_sizes; (void)n_in; (void)out_size; (void)ws_size;
  const float* x   = (const float*)d_in[0];
  const int*   ei  = (const int*)d_in[1];
  // d_in[2] lv_group_ids: provably dead (lv mask is never 0 -> logits unchanged)
  const float* W1  = (const float*)d_in[3];
  const float* b1  = (const float*)d_in[4];
  const float* W2  = (const float*)d_in[5];
  const float* b2  = (const float*)d_in[6];
  const float* W3  = (const float*)d_in[7];
  const float* b3  = (const float*)d_in[8];
  const float* We1 = (const float*)d_in[9];
  const float* be1 = (const float*)d_in[10];
  const float* We2 = (const float*)d_in[11];
  const float* be2 = (const float*)d_in[12];
  float* outF = (float*)d_out;

  char* wp = (char*)d_ws;
  auto take = [&](size_t bytes) -> void* {
    void* r = (void*)wp;
    wp += (bytes + 255) & ~(size_t)255;
    return r;
  };
  __bf16* x16   = (__bf16*)take((size_t)NN * DD * 2);
  __bf16* W1T   = (__bf16*)take((size_t)128 * 256 * 2);   // [N x K]
  __bf16* W2T   = (__bf16*)take((size_t)64 * 128 * 2);
  __bf16* W3T   = (__bf16*)take((size_t)32 * 64 * 2);
  __bf16* We1T  = (__bf16*)take((size_t)256 * 256 * 2);
  __bf16* We2T  = (__bf16*)take((size_t)256 * 256 * 2);
  __bf16* h1b   = (__bf16*)take((size_t)NN * 128 * 2);
  __bf16* h2b   = (__bf16*)take((size_t)NN * 64 * 2);
  float*  logits = (float*)take((size_t)NN * KP * 4);
  float*  Ssoft  = (float*)take((size_t)NN * KP * 4);
  float*  Sf     = (float*)take((size_t)NN * KP * 4);
  float*  Sadj   = (float*)take((size_t)NN * KP * 4);
  float*  Scur   = (float*)take((size_t)NN * KP * 4);
  __bf16* e1b   = (__bf16*)take((size_t)NN * 256 * 2);
  __bf16* xebT  = (__bf16*)take((size_t)256 * NN * 2);    // x_embed transposed
  __bf16* S16   = (__bf16*)take((size_t)NN * KP * 2);
  __bf16* S16T  = (__bf16*)take((size_t)KP * NN * 2);     // S transposed
  float* cs1 = (float*)take(32 * 4);
  float* csS = (float*)take(32 * 4);
  float* cs2 = (float*)take(32 * 4);
  float* csF = (float*)take(32 * 4);
  float* pen = (float*)take(32 * 4);
  int*   vo  = (int*)take(32 * 4);
  int*   flg = (int*)take(4);
  float* tileP = (float*)take((size_t)(NN/32) * (NN/16) * 4);
  float* edgeT = (float*)take((size_t)EE * 4);
  float* blkAdj = (float*)take((size_t)(EE / 256) * (KK * KK) * 4);
  unsigned* mask = (unsigned*)take((size_t)NN * NN / 8);
  float* sc = (float*)take(16 * 4);

  const int T = 256;
  auto cdiv = [](int a, int b) { return (a + b - 1) / b; };

  // --- bf16 conversions; weights transposed to [N x K] for contiguous frags
  k_cvt<<<cdiv(NN * DD, T), T, 0, stream>>>(x, x16, NN, DD, DD);
  k_cvtT<<<cdiv(128 * 256, T), T, 0, stream>>>(W1, W1T, 256, 128, 128);
  k_cvtT<<<cdiv(64 * 128, T), T, 0, stream>>>(W2, W2T, 128, 64, 64);
  k_cvtT<<<cdiv(32 * 64, T), T, 0, stream>>>(W3, W3T, 64, 20, 32);
  k_cvtT<<<cdiv(256 * 256, T), T, 0, stream>>>(We1, We1T, 256, 256, 256);
  k_cvtT<<<cdiv(256 * 256, T), T, 0, stream>>>(We2, We2T, 256, 256, 256);

  // --- MLP: h1 = relu(x@W1+b1), h2 = relu(h1@W2+b2), logits = h2@W3+b3 ---
  k_wmma_gemm<256, 256, 256, 128, 1><<<dim3(128/32, NN/16), 32, 0, stream>>>(
      x16, W1T, b1, nullptr, h1b, 0, 0, NN, 128, 128);
  k_wmma_gemm<128, 128, 128, 64, 1><<<dim3(64/32, NN/16), 32, 0, stream>>>(
      h1b, W2T, b2, nullptr, h2b, 0, 0, NN, 64, 64);
  k_wmma_gemm<64, 64, 64, 32, 0><<<dim3(32/32, NN/16), 32, 0, stream>>>(
      h2b, W3T, b3, logits, nullptr, 0, 0, NN, 32, 20);

  // --- softmax -> colsum feedback -> softmax ---
  k_softmax_rows<<<cdiv(NN, T), T, 0, stream>>>(logits, nullptr, Ssoft);
  k_colsum<<<KK, 256, 0, stream>>>(Ssoft, cs1);
  k_softmax_rows<<<cdiv(NN, T), T, 0, stream>>>(logits, cs1, Sf);

  // --- size constraints ---
  k_colsum<<<KK, 256, 0, stream>>>(Sf, csS);
  k_pen<<<1, 32, 0, stream>>>(csS, pen, vo, flg);
  k_sadj<<<cdiv(NN, T), T, 0, stream>>>(Sf, pen, Sadj);
  for (int k = 0; k < KK; ++k)
    k_quant_zero<<<1, 1024, 0, stream>>>(Sadj, vo, k);
  k_select<<<cdiv(NN * KP, T), T, 0, stream>>>(Sadj, Sf, flg, Scur, NN * KP);

  // --- hard rebalance: 5 sweeps over 20 columns ---
  for (int it = 0; it < 5; ++it) {
    k_colsum<<<KK, 256, 0, stream>>>(Scur, cs2);
    for (int k = 0; k < KK; ++k)
      k_rebalance<<<1, 1024, 0, stream>>>(Scur, cs2, k);
  }
  k_cvtS<<<cdiv(NN * KP, T), T, 0, stream>>>(Scur, S16, S16T);

  // --- embed path; x_embed written transposed for the pooling GEMM --------
  k_wmma_gemm<256, 256, 256, 256, 1><<<dim3(256/32, NN/16), 32, 0, stream>>>(
      x16, We1T, be1, nullptr, e1b, 0, 0, NN, 256, 256);
  k_wmma_gemm<256, 256, 256, 256, 0><<<dim3(256/32, NN/16), 32, 0, stream>>>(
      e1b, We2T, be2, nullptr, xebT, 1, NN, NN, 256, 256);
  // x_pooled = S^T @ x_embed : A = S16T [32 x NN], BT = xebT [256 x NN]
  k_wmma_gemm<NN, NN, NN, 256, 0><<<dim3(256/32, 2), 32, 0, stream>>>(
      S16T, xebT, nullptr, outF, nullptr, 0, 0, KK, 256, 256);

  // --- edges: dedup bitmask, adj_pooled, edge loss corrections ---
  k_zero32<<<cdiv(NN * NN / 32, T), T, 0, stream>>>(mask, NN * NN / 32);
  k_edges<<<EE / 256, 256, 0, stream>>>(ei, Scur, mask, edgeT, blkAdj);
  k_adj_reduce<<<1, 512, 0, stream>>>(blkAdj, outF + KK * DD);

  // --- N^2 link-loss tiles via WMMA (symmetry: ~half the tiles do work) ---
  k_ptile<<<dim3(NN/32, NN/16), 32, 0, stream>>>(S16, tileP);
  k_reduce_sum<<<1, 1024, 0, stream>>>(tileP, (NN/32) * (NN/16), sc + 0);
  k_reduce_sum<<<1, 1024, 0, stream>>>(edgeT, EE, sc + 1);
  k_entropy<<<1, 1024, 0, stream>>>(Scur, sc + 2);
  k_colsum<<<KK, 256, 0, stream>>>(Scur, csF);

  // --- outputs: S and aux ---
  k_scopy<<<cdiv(NN * KK, T), T, 0, stream>>>(Scur, outF + KK * DD + KK * KK);
  k_final<<<1, 32, 0, stream>>>(sc, csF, outF + KK * DD + KK * KK + NN * KK);
}
